// GraphConvolution_3908420239433
// MI455X (gfx1250) — compile-verified, matrix operand order checked by reading, not run on hardware
//
#include <hip/hip_runtime.h>

// Problem constants (match reference)
#define N_NODES 100000
#define N_COMM  1024
#define N_EDGES 3200000
#define D       128                 // D_IN == D_OUT
#define LDT     (N_NODES + N_COMM)  // 101024, leading dim of transposed support

#define KSPLIT  25
#define KCHUNK  (N_NODES / KSPLIT)  // 4000, multiple of 32
#define G1_ITERS (N_COMM / 32)      // 32
#define G2_ITERS (KCHUNK / 32)      // 125

typedef __attribute__((ext_vector_type(16))) __bf16 v16bf;
typedef __attribute__((ext_vector_type(8)))  __bf16 v8bf;
typedef __attribute__((ext_vector_type(8)))  float  v8f;
typedef __attribute__((ext_vector_type(4)))  unsigned int v4u;
typedef __attribute__((ext_vector_type(8)))  int    v8i;
typedef __attribute__((ext_vector_type(4)))  int    v4i;

// The TDM engine writes LDS outside the compiler's memory model.  This opaque
// barrier makes the compiler assume memory (incl. LDS) may have been written,
// so the subsequent ds_load fragments cannot be folded to poison/DCE'd.
__device__ __forceinline__ void opaque_mem_writer() {
    asm volatile("" ::: "memory");
}

// K index within a 16x32 bf16 WMMA A/B^T fragment for element e (0..15),
// lane-half h (0: lanes 0-15, 1: lanes 16-31).  (ISA 7.12.2, 16-bit A 16x32.)
// Note: e=0..7  -> K = 8h + e        (contiguous run 1)
//       e=8..15 -> K = 16 + 8h + e-8 (contiguous run 2)
__device__ __forceinline__ int frag_k(int e, int h) {
    int v = e >> 1;
    int j = e & 1;
    return 16 * (v >> 2) + 8 * h + 2 * (v & 3) + j;
}

// ---------------------------------------------------------------------------
// Tensor Data Mover: 2D tile (f32) global -> LDS.  D# built per ISA ch.8:
// group0: count=1, lds_addr, global_addr, type=2; group1: data_size=4B,
// tensor dims, tile dims, dim0 stride.  Groups 2/3 zero (2D tensor).
// ---------------------------------------------------------------------------
__device__ __forceinline__ void tdm_load_2d(unsigned lds_off, const void* gptr,
                                            unsigned tile_x, unsigned tile_y,
                                            unsigned tdim0, unsigned tdim1,
                                            unsigned stride0_elems) {
    unsigned long long ga = (unsigned long long)(uintptr_t)gptr;
    v4u g0;
    g0[0] = 1u;                                               // count=1, user D#
    g0[1] = lds_off;                                          // lds_addr (bytes)
    g0[2] = (unsigned)(ga & 0xFFFFFFFFu);                     // global_addr[31:0]
    g0[3] = (unsigned)((ga >> 32) & 0x1FFFFFFu) | (2u << 30); // addr[56:32], type=2
    v8i g1;
    g1[0] = (int)(2u << 16);                                  // data_size=2 (4B)
    g1[1] = (int)((tdim0 & 0xFFFFu) << 16);                   // tensor_dim0[15:0]
    g1[2] = (int)((tdim0 >> 16) | ((tdim1 & 0xFFFFu) << 16)); // dim0 hi | dim1 lo
    g1[3] = (int)((tdim1 >> 16) | (tile_x << 16));            // dim1 hi | tile_dim0
    g1[4] = (int)(tile_y & 0xFFFFu);                          // tile_dim1, tile_dim2=0
    g1[5] = (int)stride0_elems;                               // tensor_dim0_stride
    g1[6] = 0;
    g1[7] = 0;
    v4i z4 = {0, 0, 0, 0};
#if defined(__clang_major__) && (__clang_major__ >= 23)
    v8i z8 = {0, 0, 0, 0, 0, 0, 0, 0};
    __builtin_amdgcn_tensor_load_to_lds(g0, g1, z4, z4, z8, 0);
#else
    __builtin_amdgcn_tensor_load_to_lds(g0, g1, z4, z4, 0);
#endif
}

// Load A fragment (16x32) from row-major f32 (global), converting to bf16.
__device__ __forceinline__ v16bf load_a_rowmajor_f32(const float* __restrict__ A, int lda,
                                                     int row, int k0, int h) {
    v16bf a;
#pragma unroll
    for (int p = 0; p < 8; ++p) {
        int kb = 16 * (p >> 2) + 8 * h + 2 * (p & 3);
        float2 f2 = *reinterpret_cast<const float2*>(A + (size_t)row * lda + k0 + kb);
        a[2 * p]     = (__bf16)f2.x;
        a[2 * p + 1] = (__bf16)f2.y;
    }
    return a;
}

// B^T fragment (column `col` of row-major KxN f32) - small weight matrix only.
__device__ __forceinline__ v16bf load_bT_f32(const float* __restrict__ B, int ldb,
                                             int col, int k0, int h) {
    v16bf b;
#pragma unroll
    for (int e = 0; e < 16; ++e)
        b[e] = (__bf16)B[(size_t)(k0 + frag_k(e, h)) * ldb + col];
    return b;
}

// A fragment from an LDS-staged 16x32 f32 tile (row-major), fused row sum.
// Two 32B contiguous runs per lane -> 4x ds_load_b128.
__device__ __forceinline__ v16bf frag_from_lds_row(const float* tile, int row, int h,
                                                   float* rsum) {
    const float4* p = (const float4*)(tile + row * 32 + 8 * h);
    const float4* q = (const float4*)(tile + row * 32 + 16 + 8 * h);
    float4 a0 = p[0], a1 = p[1], b0 = q[0], b1 = q[1];
    *rsum += a0.x + a0.y + a0.z + a0.w + a1.x + a1.y + a1.z + a1.w +
             b0.x + b0.y + b0.z + b0.w + b1.x + b1.y + b1.z + b1.w;
    v16bf a;
    a[0] = (__bf16)a0.x;  a[1] = (__bf16)a0.y;  a[2] = (__bf16)a0.z;  a[3] = (__bf16)a0.w;
    a[4] = (__bf16)a1.x;  a[5] = (__bf16)a1.y;  a[6] = (__bf16)a1.z;  a[7] = (__bf16)a1.w;
    a[8] = (__bf16)b0.x;  a[9] = (__bf16)b0.y;  a[10] = (__bf16)b0.z; a[11] = (__bf16)b0.w;
    a[12] = (__bf16)b1.x; a[13] = (__bf16)b1.y; a[14] = (__bf16)b1.z; a[15] = (__bf16)b1.w;
    return a;
}

// B^T fragment from the transposed bf16 support (contiguous in K):
// two 16B global_load_b128 per fragment.
__device__ __forceinline__ v16bf frag_from_supT(const __bf16* __restrict__ col_ptr,
                                                int k0, int h) {
    v8bf lo = *(const v8bf*)(col_ptr + k0 + 8 * h);
    v8bf hi = *(const v8bf*)(col_ptr + k0 + 16 + 8 * h);
    return __builtin_shufflevector(lo, hi, 0, 1, 2, 3, 4, 5, 6, 7,
                                   8, 9, 10, 11, 12, 13, 14, 15);
}

// ---------------------------------------------------------------------------
// Kernel 1: support = x @ W -> bf16 workspace, written twice:
// row-major `sup` (for the SpMM gather) and transposed `supT` (B operand).
// One wave per 16x16 tile; K = 128.
// ---------------------------------------------------------------------------
__global__ void gcn_support_gemm(const float* __restrict__ x, const float* __restrict__ w,
                                 __bf16* __restrict__ sup, __bf16* __restrict__ supT) {
    const int wave = (blockIdx.x * blockDim.x + threadIdx.x) >> 5;
    const int lane = threadIdx.x & 31;
    const int tn = wave & 7;           // 8 tiles over N=128
    const int tm = wave >> 3;          // 6314 tiles over M=101024
    const int h   = lane >> 4;
    const int l16 = lane & 15;
    const int row = tm * 16 + l16;

    v8f acc = {};
#pragma unroll
    for (int k0 = 0; k0 < D; k0 += 32) {
        v16bf a = load_a_rowmajor_f32(x, D, row, k0, h);
        v16bf b = load_bT_f32(w, D, tn * 16 + l16, k0, h);
        acc = __builtin_amdgcn_wmma_f32_16x16x32_bf16(false, a, false, b,
                                                      (short)0, acc, false, false);
    }
#pragma unroll
    for (int v = 0; v < 8; ++v) {
        int r = tm * 16 + v + 8 * h;   // C/D layout: VGPR v -> row v (+8 for hi lanes)
        __bf16 val = (__bf16)acc[v];
        sup [(size_t)r * D + tn * 16 + l16]   = val;
        supT[(size_t)(tn * 16 + l16) * LDT + r] = val;
    }
}

// ---------------------------------------------------------------------------
// Kernel 2: out[0:N,:] = (assignments / rowsum) @ sup_comm.
// Block = 8 waves = one 16-row strip x all 128 cols.  The shared 16x32 f32
// A-tile is TDM-staged into LDS (double-buffered, wave 0 drives the DMA).
// ---------------------------------------------------------------------------
__global__ void gcn_comm_to_node(const float* __restrict__ assign,
                                 const __bf16* __restrict__ supT,
                                 float* __restrict__ out) {
    __shared__ float atile[2][16 * 32];
    const int lane = threadIdx.x & 31;
    const int tn   = threadIdx.x >> 5;   // 0..7
    const int tm   = blockIdx.x;         // 0..6249
    const int h    = lane >> 4;
    const int l16  = lane & 15;

    // Real stores so the compiler knows atile is written (TDM overwrites it).
    for (int i = threadIdx.x; i < 2 * 16 * 32; i += blockDim.x)
        (&atile[0][0])[i] = 0.f;
    __syncthreads();

    const float*  gA   = assign + (size_t)tm * 16 * N_COMM;
    const __bf16* bcol = supT + (size_t)(tn * 16 + l16) * LDT + N_NODES;

    if (threadIdx.x < 32)
        tdm_load_2d(0u, gA, 32, 16, N_COMM, N_NODES, N_COMM);

    v8f acc = {};
    float rsum = 0.f;
    for (int it = 0; it < G1_ITERS; ++it) {
        if (threadIdx.x < 32) {
            if (it + 1 < G1_ITERS) {
                tdm_load_2d((unsigned)(((it + 1) & 1) * 16 * 32 * sizeof(float)),
                            gA + (it + 1) * 32, 32, 16, N_COMM, N_NODES, N_COMM);
                __builtin_amdgcn_s_wait_tensorcnt(1);   // current buffer ready
            } else {
                __builtin_amdgcn_s_wait_tensorcnt(0);
            }
        }
        __syncthreads();
        opaque_mem_writer();                 // LDS was filled by the TDM
        v16bf a = frag_from_lds_row(atile[it & 1], l16, h, &rsum);
        v16bf b = frag_from_supT(bcol, it * 32, h);
        acc = __builtin_amdgcn_wmma_f32_16x16x32_bf16(false, a, false, b,
                                                      (short)0, acc, false, false);
        __syncthreads();
    }
    rsum += __shfl_xor(rsum, 16, 32);    // full row sum for row tm*16 + l16
#pragma unroll
    for (int v = 0; v < 8; ++v) {
        int rloc = v + 8 * h;
        float denom = __shfl(rsum, rloc, 32);
        out[(size_t)(tm * 16 + rloc) * D + tn * 16 + l16] = acc[v] / denom;
    }
}

// ---------------------------------------------------------------------------
// Kernel 3: SpMM scatter: out[row] += val * sup_nodes[col].  L2-resident
// gather (bf16 rows) + f32 atomic adds; one wave per edge iteration.
// ---------------------------------------------------------------------------
__global__ void gcn_spmm_edges(const int* __restrict__ erow, const int* __restrict__ ecol,
                               const float* __restrict__ eval,
                               const __bf16* __restrict__ sup_nodes,
                               float* __restrict__ out) {
    const int lane   = threadIdx.x & 31;
    const int wave   = (blockIdx.x * blockDim.x + threadIdx.x) >> 5;
    const int nwaves = (gridDim.x * blockDim.x) >> 5;
    for (int e = wave; e < N_EDGES; e += nwaves) {
        __builtin_prefetch(erow + e + nwaves, 0, 1);
        int   r = erow[e];
        int   c = ecol[e];
        float v = eval[e];
        const __bf16* src = sup_nodes + (size_t)c * D + lane * 4;
        float*        dst = out       + (size_t)r * D + lane * 4;
#pragma unroll
        for (int j = 0; j < 4; ++j)
            atomicAdd(dst + j, v * (float)src[j]);
    }
}

// ---------------------------------------------------------------------------
// Kernel 4: out2 += (A^T K-chunk) @ sup_nodes (split-K atomics) + colsums.
// Block = one (community-tile, K-chunk); 8 waves = 8 N tiles share the
// TDM-staged 32x16 f32 A^T tile in LDS (double-buffered).
// ---------------------------------------------------------------------------
__global__ void gcn_at_gemm(const float* __restrict__ assign,
                            const __bf16* __restrict__ supT,
                            float* __restrict__ out2, float* __restrict__ col_sums) {
    __shared__ float atile[2][32 * 16];
    const int lane = threadIdx.x & 31;
    const int tn   = threadIdx.x >> 5;   // 0..7
    const int tm   = blockIdx.x;         // 0..63 community tiles
    const int ks   = blockIdx.y;         // 0..KSPLIT-1
    const int h    = lane >> 4;
    const int l16  = lane & 15;
    const int kbeg = ks * KCHUNK;

    for (int i = threadIdx.x; i < 2 * 32 * 16; i += blockDim.x)
        (&atile[0][0])[i] = 0.f;
    __syncthreads();

    const float*  gA   = assign + (size_t)kbeg * N_COMM + tm * 16;
    const __bf16* bcol = supT + (size_t)(tn * 16 + l16) * LDT;

    if (threadIdx.x < 32)
        tdm_load_2d(0u, gA, 16, 32, N_COMM, N_NODES, N_COMM);

    v8f acc = {};
    float csum = 0.f;
    for (int it = 0; it < G2_ITERS; ++it) {
        if (threadIdx.x < 32) {
            if (it + 1 < G2_ITERS) {
                tdm_load_2d((unsigned)(((it + 1) & 1) * 32 * 16 * sizeof(float)),
                            gA + (size_t)(it + 1) * 32 * N_COMM,
                            16, 32, N_COMM, N_NODES, N_COMM);
                __builtin_amdgcn_s_wait_tensorcnt(1);
            } else {
                __builtin_amdgcn_s_wait_tensorcnt(0);
            }
        }
        __syncthreads();
        opaque_mem_writer();                   // LDS was filled by the TDM
        const float* tile = atile[it & 1];     // [k_local][community] = k*16 + c
        v16bf a;
#pragma unroll
        for (int e = 0; e < 16; ++e) {
            float f = tile[frag_k(e, h) * 16 + l16];
            csum += f;
            a[e] = (__bf16)f;
        }
        v16bf b = frag_from_supT(bcol, kbeg + it * 32, h);
        acc = __builtin_amdgcn_wmma_f32_16x16x32_bf16(false, a, false, b,
                                                      (short)0, acc, false, false);
        __syncthreads();
    }
    csum += __shfl_xor(csum, 16, 32);
    if (tn == 0 && lane < 16)
        atomicAdd(col_sums + tm * 16 + l16, csum);
#pragma unroll
    for (int v = 0; v < 8; ++v) {
        int rloc = v + 8 * h;
        atomicAdd(out2 + (size_t)(tm * 16 + rloc) * D + tn * 16 + l16, acc[v]);
    }
}

// ---------------------------------------------------------------------------
// Kernel 5: out2 /= colsum (per community row).
// ---------------------------------------------------------------------------
__global__ void gcn_finalize(float* __restrict__ out2, const float* __restrict__ col_sums) {
    int idx = blockIdx.x * blockDim.x + threadIdx.x;   // 0 .. 1024*128-1
    int c = idx >> 7;
    out2[idx] = out2[idx] / col_sums[c];
}

extern "C" void kernel_launch(void* const* d_in, const int* in_sizes, int n_in,
                              void* d_out, int out_size, void* d_ws, size_t ws_size,
                              hipStream_t stream) {
    const float* x      = (const float*)d_in[0];
    const float* w      = (const float*)d_in[1];
    const float* assign = (const float*)d_in[2];
    const int*   erow   = (const int*)d_in[3];
    const int*   ecol   = (const int*)d_in[4];
    const float* eval   = (const float*)d_in[5];

    float* out  = (float*)d_out;
    float* out2 = out + (size_t)N_NODES * D;

    const size_t sup_bytes = (size_t)LDT * D * sizeof(__bf16);   // 25,862,144
    __bf16* sup      = (__bf16*)d_ws;                            // row-major (LDT x 128)
    __bf16* supT     = (__bf16*)((char*)d_ws + sup_bytes);       // transposed (128 x LDT)
    float*  col_sums = (float*)((char*)d_ws + 2 * sup_bytes);

    // Zero accumulation targets (graph-capture-safe stream ops).
    hipMemsetAsync(out2, 0, (size_t)N_COMM * D * sizeof(float), stream);
    hipMemsetAsync(col_sums, 0, (size_t)N_COMM * sizeof(float), stream);

    // 1) support = x @ W  (bf16, row-major + transposed)
    {
        int waves = (LDT / 16) * (D / 16);                   // 50512
        gcn_support_gemm<<<waves * 32 / 256, 256, 0, stream>>>(x, w, sup, supT);
    }
    // 2) d_out[0:N] = a_norm @ sup_comm   (TDM-staged A tiles)
    gcn_comm_to_node<<<N_NODES / 16, 256, 0, stream>>>(assign, supT, out);
    // 3) d_out[0:N] += segment_sum(edge_val * sup_nodes[col])
    gcn_spmm_edges<<<8192, 256, 0, stream>>>(erow, ecol, eval, sup, out);
    // 4) d_out[N:] = at_norm @ sup_nodes (split-K accumulate, TDM-staged A^T)
    gcn_at_gemm<<<dim3(N_COMM / 16, KSPLIT), 256, 0, stream>>>(assign, supT, out2, col_sums);
    // 5) normalize out2 rows
    gcn_finalize<<<(N_COMM * D) / 256, 256, 0, stream>>>(out2, col_sums);
}
